// FFTConv1d_70257075028063
// MI455X (gfx1250) — matrix operand order repaired and checked by reading, not compiled, not run on hardware
//
#include <hip/hip_runtime.h>

typedef __attribute__((ext_vector_type(16))) _Float16 v16h;
typedef __attribute__((ext_vector_type(8)))  float    v8f;
typedef __attribute__((ext_vector_type(4)))  float    f32x4;

#define B_    8
#define CIN_  64
#define COUT_ 64
#define T_    16384
#define K_    127
#define HALO  63              // (K-1)/2, centered 'same' crop
#define TN    256             // t-tile per workgroup
#define XROWS (TN + K_ - 1)   // 382 staged rows (tile + halo)
#define PITCH 72              // f16 per LDS row (64 + 8 pad; 144B = 9*16B aligned)

// ---------------------------------------------------------------------------
// Repack W[cout][cin][k] (f32) into WMMA A-fragment-ready f16 with the kernel
// flipped (126-k), so the main loop is y[t] += W'[k] * x[t-63+k].
// Layout: [k][cb:4][ch:2][lane:32][e:16]  (flat, 16 f16 = 32B per lane).
// A 16x32 f16 fragment (ISA 7.12.2): lane l -> row M=l%16; half h=l/16:
//   h=0: e0..7 -> K=0..7,  e8..15 -> K=16..23
//   h=1: e0..7 -> K=8..15, e8..15 -> K=24..31
// ---------------------------------------------------------------------------
__global__ __launch_bounds__(256) void pack_w(const float* __restrict__ W,
                                              _Float16* __restrict__ Wf) {
  int f = blockIdx.x * 256 + threadIdx.x;
  int k = f >> 12;                  // 4096 elements per k
  if (k >= K_) return;
  int e    = f & 15;
  int lane = (f >> 4) & 31;
  int ch   = (f >> 9) & 1;
  int cb   = (f >> 10) & 3;
  int h    = lane >> 4;
  int cout = cb * 16 + (lane & 15);
  int cin  = ch * 32 + (e & 7) + h * 8 + ((e >> 3) << 4);
  float w  = W[((size_t)cout * CIN_ + cin) * K_ + (K_ - 1 - k)];
  Wf[f] = (_Float16)w;
}

// ---------------------------------------------------------------------------
// Transpose + downconvert x[b][cin][t] (f32) -> Xt[b][t][cin] (f16) so conv
// tiles are contiguous 128B rows (async-to-LDS friendly). 64x64 LDS tile.
// ---------------------------------------------------------------------------
__global__ __launch_bounds__(256) void pack_x(const float* __restrict__ x,
                                              _Float16* __restrict__ Xt) {
  __shared__ float Ls[64][65];
  const int b   = blockIdx.y;
  const int t0v = blockIdx.x * 64;
  const int a   = threadIdx.x >> 6;   // 0..3
  const int l   = threadIdx.x & 63;   // 0..63
#pragma unroll
  for (int i = 0; i < 16; ++i) {      // coalesced reads along t
    int cin = i * 4 + a;
    Ls[cin][l] = x[((size_t)b * CIN_ + cin) * T_ + t0v + l];
  }
  __syncthreads();
#pragma unroll
  for (int i = 0; i < 16; ++i) {      // coalesced writes along cin
    int t = i * 4 + a;
    Xt[((size_t)b * T_ + t0v + t) * CIN_ + l] = (_Float16)Ls[l][t];
  }
}

// ---------------------------------------------------------------------------
// Implicit-GEMM conv: one workgroup = 64 couts x 256 t-positions for one b.
// 8 waves: wave w -> cout half mh=w&1 (2 x 16-row tiles), t quarter th=w>>1
// (4 x 16-col tiles). Each B fragment feeds 2 WMMAs (DS bytes/wmma halved).
// Staging uses CDNA5 async global->LDS (ASYNCcnt) from pre-packed f16 x.
// ---------------------------------------------------------------------------
__global__ __launch_bounds__(256) void conv_wmma(const _Float16* __restrict__ Xt,
                                                 const _Float16* __restrict__ Wf,
                                                 float* __restrict__ y) {
  __shared__ _Float16 Xs[XROWS * PITCH];   // Xs[t_local][cin], ~55KB

  const int b   = blockIdx.y;
  const int t0  = blockIdx.x * TN;
  const int tid = threadIdx.x;
  const _Float16* xtb = Xt + (size_t)b * T_ * CIN_;

  // Stage 382 rows x 128B via async copy; zero-fill halo rows outside [0,T).
#pragma unroll
  for (int it = 0; it < 12; ++it) {
    int id  = it * 256 + tid;           // 382*8 = 3056 16B-chunks
    int row = id >> 3;
    int c   = id & 7;
    if (row < XROWS) {
      int tg = t0 - HALO + row;
      _Float16* lp = &Xs[row * PITCH + c * 8];
      if (tg >= 0 && tg < T_) {
        const _Float16* gp = xtb + ((size_t)tg << 6) + c * 8;
        unsigned loff = (unsigned)(unsigned long long)lp;  // low 32b = LDS offset
        asm volatile("global_load_async_to_lds_b128 %0, %1, off"
                     :: "v"(loff), "v"(gp) : "memory");
      } else {
        f32x4 z = {0.f, 0.f, 0.f, 0.f};
        *(f32x4*)lp = z;
      }
    }
  }
  asm volatile("s_wait_asynccnt 0x0" ::: "memory");
  __syncthreads();

  const int lane = tid & 31;
  const int wv   = tid >> 5;
  const int mh   = wv & 1;       // cout half (2 x 16 rows)
  const int th   = wv >> 1;      // t quarter (64 cols)
  const int lm   = lane & 15;
  const int hh   = lane >> 4;

  v8f acc[2][4] = {};            // 2 cout-tiles x 4 t-tiles of 16x16 f32

  // A-fragment base; [k][cb][ch][lane][16], per-k stride = 4096 f16.
  const _Float16* __restrict__ wf0 = Wf + (size_t)(mh * 2) * 1024 + (size_t)lane * 16;
  const int xbase = (th * 64 + lm) * PITCH + hh * 8;

  for (int k = 0; k < K_; ++k) {
    const _Float16* wk = wf0 + (size_t)k * 4096;
    const int xrow = xbase + k * PITCH;
#pragma unroll
    for (int ch = 0; ch < 2; ++ch) {
      union { v16h v; f32x4 q[2]; } a0, a1;
      const f32x4* ap0 = (const f32x4*)(wk + ch * 512);
      const f32x4* ap1 = (const f32x4*)(wk + 1024 + ch * 512);
      a0.q[0] = ap0[0]; a0.q[1] = ap0[1];
      a1.q[0] = ap1[0]; a1.q[1] = ap1[1];
#pragma unroll
      for (int j = 0; j < 4; ++j) {
        // B fragment: cin 8h..8h+7 and 16+8h..16+8h+7 per lane -> 2x ds_load_b128.
        union { v16h v; f32x4 q[2]; } bu;
        const _Float16* bp = &Xs[xrow + j * (16 * PITCH) + ch * 32];
        bu.q[0] = *(const f32x4*)bp;
        bu.q[1] = *(const f32x4*)(bp + 16);
        acc[0][j] = __builtin_amdgcn_wmma_f32_16x16x32_f16(
            false, a0.v, false, bu.v, (short)0, acc[0][j], false, false);
        acc[1][j] = __builtin_amdgcn_wmma_f32_16x16x32_f16(
            false, a1.v, false, bu.v, (short)0, acc[1][j], false, false);
      }
    }
  }

  // C/D 16x16 f32 layout: lane l -> N=l%16, VGPR v -> M=(l/16)*8 + v.
  const int tb = t0 + th * 64 + lm;
#pragma unroll
  for (int m = 0; m < 2; ++m) {
    const size_t cbase = (size_t)b * COUT_ + (size_t)((mh * 2 + m) * 16 + hh * 8);
#pragma unroll
    for (int j = 0; j < 4; ++j) {
#pragma unroll
      for (int v = 0; v < 8; ++v) {
        y[(cbase + v) * T_ + tb + j * 16] = acc[m][j][v];
      }
    }
  }
}

extern "C" void kernel_launch(void* const* d_in, const int* in_sizes, int n_in,
                              void* d_out, int out_size, void* d_ws, size_t ws_size,
                              hipStream_t stream) {
  const float* x = (const float*)d_in[0];   // [B, CIN, T] f32
  const float* W = (const float*)d_in[1];   // [COUT, CIN, K] f32
  float* y = (float*)d_out;                 // [B, COUT, T] f32

  _Float16* Wf = (_Float16*)d_ws;                           // 1.04 MB packed W
  _Float16* Xt = (_Float16*)((char*)d_ws + (1u << 20));     // 16.8 MB packed x

  const int packElems = K_ * 4096;          // 520192
  pack_w<<<dim3((packElems + 255) / 256), 256, 0, stream>>>(W, Wf);
  pack_x<<<dim3(T_ / 64, B_), 256, 0, stream>>>(x, Xt);
  conv_wmma<<<dim3(T_ / TN, B_), 256, 0, stream>>>(Xt, Wf, y);
}
